// L2Model_58059367908120
// MI455X (gfx1250) — matrix operand order repaired and checked by reading, not compiled, not run on hardware
//
#include <hip/hip_runtime.h>

#define N_ROW 8192
#define N_COL 8192
#define RANK  256

#define BM 128
#define BN 128
#define BK 32
#define LSTR 40   // LDS row stride in bf16 elements (80B) -> conflict-free b128 column reads
#define KSTEPS (RANK / BK)

typedef __bf16 bf16_t;
typedef __attribute__((ext_vector_type(16))) bf16_t        v16bf;
typedef __attribute__((ext_vector_type(8)))  float         v8f;
typedef __attribute__((ext_vector_type(8)))  unsigned short u16x8;

union FragCast {
    u16x8 u[2];
    v16bf b;
};

static __device__ __forceinline__ unsigned short f32_to_bf16(float f) {
    unsigned int u = __float_as_uint(f);
    unsigned int r = u + 0x7FFFu + ((u >> 16) & 1u);   // round-to-nearest-even
    return (unsigned short)(r >> 16);
}
static __device__ __forceinline__ float bf16_to_f32(unsigned short h) {
    return __uint_as_float(((unsigned int)h) << 16);
}

// CDNA5 async DMA: global -> LDS, 16B per lane, tracked by ASYNCcnt (GVS mode).
static __device__ __forceinline__ void async_copy_b128(unsigned lds_byte,
                                                       const unsigned short* base,
                                                       int gbyte) {
    asm volatile("global_load_async_to_lds_b128 %0, %1, %2"
                 :: "v"(lds_byte), "v"(gbyte), "s"(base)
                 : "memory");
}
static __device__ __forceinline__ void wait_async0() {
    asm volatile("s_wait_asynccnt 0x0" ::: "memory");
}

// --- prep 1: split X rows into bf16 hi/lo, compute exact f32 row norms ---
__global__ void split_x_kernel(const float* __restrict__ X,
                               unsigned short* __restrict__ Xh,
                               unsigned short* __restrict__ Xl,
                               float* __restrict__ x2) {
    __shared__ float red[256];
    const int row = blockIdx.x;
    const int k   = threadIdx.x;           // RANK == 256 == blockDim.x
    const float v = X[(size_t)row * RANK + k];
    const unsigned short h = f32_to_bf16(v);
    const unsigned short l = f32_to_bf16(v - bf16_to_f32(h));
    Xh[(size_t)row * RANK + k] = h;
    Xl[(size_t)row * RANK + k] = l;
    red[k] = v * v;
    __syncthreads();
    #pragma unroll
    for (int s = 128; s > 0; s >>= 1) {
        if (k < s) red[k] += red[k + s];
        __syncthreads();
    }
    if (k == 0) x2[row] = red[0];
}

// --- prep 2: exact f32 column norms of Y (coalesced along n) ---
__global__ void colnorm_y_kernel(const float* __restrict__ Y,
                                 float* __restrict__ y2) {
    const int n = blockIdx.x * blockDim.x + threadIdx.x;
    float s = 0.0f;
    #pragma unroll 4
    for (int k = 0; k < RANK; ++k) {
        const float v = Y[(size_t)k * N_COL + n];
        s += v * v;
    }
    y2[n] = s;
}

// --- prep 3: transpose Y (K x N) -> Yt (N x K) as bf16 hi/lo via LDS tile ---
__global__ void split_yt_kernel(const float* __restrict__ Y,
                                unsigned short* __restrict__ Yth,
                                unsigned short* __restrict__ Ytl) {
    __shared__ float tile[64][65];
    const int n0 = blockIdx.x * 64;
    const int k0 = blockIdx.y * 64;
    const int tx = threadIdx.x & 63;
    const int ty = threadIdx.x >> 6;       // 0..3
    #pragma unroll
    for (int r = 0; r < 16; ++r) {
        const int kl = ty + r * 4;
        tile[kl][tx] = Y[(size_t)(k0 + kl) * N_COL + (n0 + tx)];
    }
    __syncthreads();
    #pragma unroll
    for (int r = 0; r < 16; ++r) {
        const int nl = ty + r * 4;
        const float v = tile[tx][nl];
        const unsigned short h = f32_to_bf16(v);
        const unsigned short l = f32_to_bf16(v - bf16_to_f32(h));
        const size_t idx = (size_t)(n0 + nl) * RANK + (k0 + tx);
        Yth[idx] = h;
        Ytl[idx] = l;
    }
}

// --- main: bf16x3 WMMA GEMM, async-to-LDS double buffered, fused L2 epilogue ---
__global__ __launch_bounds__(256)
void l2_wmma_kernel(const unsigned short* __restrict__ Xh,
                    const unsigned short* __restrict__ Xl,
                    const unsigned short* __restrict__ Yh,
                    const unsigned short* __restrict__ Yl,
                    const float* __restrict__ x2,
                    const float* __restrict__ y2,
                    const float* __restrict__ betap,
                    float* __restrict__ out) {
    __shared__ __align__(16) unsigned short sXh[2][BM * LSTR];
    __shared__ __align__(16) unsigned short sXl[2][BM * LSTR];
    __shared__ __align__(16) unsigned short sYh[2][BN * LSTR];
    __shared__ __align__(16) unsigned short sYl[2][BN * LSTR];

    const int tid  = threadIdx.x;
    const int lane = tid & 31;
    const int wave = tid >> 5;         // 0..7
    const int wr   = wave >> 1;        // 0..3 -> 32 rows each
    const int wc   = wave & 1;         // 0..1 -> 64 cols each
    const int lr   = lane & 15;
    const int lh   = lane >> 4;

    const int rowBase = blockIdx.y * BM;
    const int colBase = blockIdx.x * BN;

    // staging: each thread DMAs 2x16B from each of the 4 matrices per K-step
    const int srow = tid >> 1;         // 0..127
    const int shc  = tid & 1;          // which 16-col half of the 32-wide chunk
    const int ldByte = (srow * LSTR + shc * 16) * 2;

    const unsigned lXh0 = (unsigned)(size_t)&sXh[0][0] + ldByte;
    const unsigned lXl0 = (unsigned)(size_t)&sXl[0][0] + ldByte;
    const unsigned lYh0 = (unsigned)(size_t)&sYh[0][0] + ldByte;
    const unsigned lYl0 = (unsigned)(size_t)&sYl[0][0] + ldByte;
    const unsigned bufStride = (unsigned)(BM * LSTR * 2);   // bytes between buf0/buf1

    const int gxRow = ((rowBase + srow) * RANK + shc * 16) * 2;  // byte offset, X-side
    const int gyRow = ((colBase + srow) * RANK + shc * 16) * 2;  // byte offset, Y-side

    auto issue_chunk = [&](int buf, int kc) {
        const int gx = gxRow + kc * 2;
        const int gy = gyRow + kc * 2;
        const unsigned b = buf ? bufStride : 0u;
        async_copy_b128(lXh0 + b,      Xh, gx);
        async_copy_b128(lXh0 + b + 16, Xh, gx + 16);
        async_copy_b128(lXl0 + b,      Xl, gx);
        async_copy_b128(lXl0 + b + 16, Xl, gx + 16);
        async_copy_b128(lYh0 + b,      Yh, gy);
        async_copy_b128(lYh0 + b + 16, Yh, gy + 16);
        async_copy_b128(lYl0 + b,      Yl, gy);
        async_copy_b128(lYl0 + b + 16, Yl, gy + 16);
    };

    v8f acc[2][4];
    #pragma unroll
    for (int i = 0; i < 2; ++i)
        #pragma unroll
        for (int j = 0; j < 4; ++j)
            acc[i][j] = (v8f){0.f, 0.f, 0.f, 0.f, 0.f, 0.f, 0.f, 0.f};

    issue_chunk(0, 0);

    int cur = 0;
    for (int kci = 0; kci < KSTEPS; ++kci) {
        wait_async0();       // this wave's DMA into buf `cur` (and any prefetch) complete
        __syncthreads();     // all waves' DMA visible; nobody still reads buf `cur^1`

        if (kci + 1 < KSTEPS)
            issue_chunk(cur ^ 1, (kci + 1) * BK);   // overlap DMA with WMMA below

        // A fragments: lane (lr,lh) holds K in {16lh..16lh+15}
        v16bf Ah[2], Al[2], Bh[4], Bl[4];
        #pragma unroll
        for (int i = 0; i < 2; ++i) {
            const int ab = (wr * 32 + i * 16 + lr) * LSTR;
            FragCast f;
            f.u[0] = *(const u16x8*)(&sXh[cur][ab + lh * 16]);
            f.u[1] = *(const u16x8*)(&sXh[cur][ab + lh * 16 + 8]);
            Ah[i] = f.b;
            f.u[0] = *(const u16x8*)(&sXl[cur][ab + lh * 16]);
            f.u[1] = *(const u16x8*)(&sXl[cur][ab + lh * 16 + 8]);
            Al[i] = f.b;
        }
        // B fragments: lane (n=lr, lh) holds K in {16lh..16lh+15}
        #pragma unroll
        for (int j = 0; j < 4; ++j) {
            const int bb = (wc * 64 + j * 16 + lr) * LSTR + lh * 16;
            FragCast f;
            f.u[0] = *(const u16x8*)(&sYh[cur][bb]);
            f.u[1] = *(const u16x8*)(&sYh[cur][bb + 8]);
            Bh[j] = f.b;
            f.u[0] = *(const u16x8*)(&sYl[cur][bb]);
            f.u[1] = *(const u16x8*)(&sYl[cur][bb + 8]);
            Bl[j] = f.b;
        }

        // bf16x3: acc += Ah*Bh + Ah*Bl + Al*Bh
        #pragma unroll
        for (int i = 0; i < 2; ++i) {
            #pragma unroll
            for (int j = 0; j < 4; ++j) {
                acc[i][j] = __builtin_amdgcn_wmma_f32_16x16x32_bf16(
                    false, Ah[i], false, Bh[j], (short)0, acc[i][j], false, false);
                acc[i][j] = __builtin_amdgcn_wmma_f32_16x16x32_bf16(
                    false, Ah[i], false, Bl[j], (short)0, acc[i][j], false, false);
                acc[i][j] = __builtin_amdgcn_wmma_f32_16x16x32_bf16(
                    false, Al[i], false, Bh[j], (short)0, acc[i][j], false, false);
            }
        }

        cur ^= 1;
    }

    // epilogue: D layout is M = vgpr + 8*lh, N = lr
    const float beta = *betap;
    #pragma unroll
    for (int i = 0; i < 2; ++i) {
        const int row0 = rowBase + wr * 32 + i * 16 + 8 * lh;
        #pragma unroll
        for (int j = 0; j < 4; ++j) {
            const int col = colBase + wc * 64 + j * 16 + lr;
            const float y2v = y2[col];
            #pragma unroll
            for (int r = 0; r < 8; ++r) {
                const int row = row0 + r;
                float d2 = x2[row] + y2v - 2.0f * acc[i][j][r];
                d2 = fmaxf(d2, 0.0f);
                out[(size_t)row * N_COL + col] = beta - __builtin_sqrtf(d2);
            }
        }
    }
}

extern "C" void kernel_launch(void* const* d_in, const int* in_sizes, int n_in,
                              void* d_out, int out_size, void* d_ws, size_t ws_size,
                              hipStream_t stream) {
    const float* X    = (const float*)d_in[0];
    const float* Y    = (const float*)d_in[1];
    const float* beta = (const float*)d_in[2];
    float* out = (float*)d_out;

    // workspace layout (bf16 splits + norms), all 256B-aligned chunks
    char* ws = (char*)d_ws;
    const size_t half = (size_t)N_ROW * RANK * sizeof(unsigned short);  // 4 MB
    unsigned short* Xh  = (unsigned short*)(ws);
    unsigned short* Xl  = (unsigned short*)(ws + half);
    unsigned short* Yth = (unsigned short*)(ws + 2 * half);
    unsigned short* Ytl = (unsigned short*)(ws + 3 * half);
    float* x2 = (float*)(ws + 4 * half);
    float* y2 = (float*)(ws + 4 * half + (size_t)N_ROW * sizeof(float));

    split_x_kernel<<<N_ROW, 256, 0, stream>>>(X, Xh, Xl, x2);
    colnorm_y_kernel<<<N_COL / 256, 256, 0, stream>>>(Y, y2);
    split_yt_kernel<<<dim3(N_COL / 64, RANK / 64), 256, 0, stream>>>(Y, Yth, Ytl);

    l2_wmma_kernel<<<dim3(N_COL / BN, N_ROW / BM), 256, 0, stream>>>(
        Xh, Xl, Yth, Ytl, x2, y2, beta, out);
}
